// NeRF_43482248905247
// MI455X (gfx1250) — matrix operand order repaired and checked by reading, not compiled, not run on hardware
//
#include <hip/hip_runtime.h>
#include <hip/hip_bf16.h>

typedef _Float16 v16h __attribute__((ext_vector_type(16)));
typedef _Float16 v8h  __attribute__((ext_vector_type(8)));
typedef float    v8f  __attribute__((ext_vector_type(8)));
typedef float    v4f  __attribute__((ext_vector_type(4)));

// ---------------- weight workspace layout (f16 elements) ----------------
// Each layer stored transposed: WT[n][k], n = output feature (padded), k padded
// to a multiple of 32 (concat layers pad the PE part to 64 with zero rows).
constexpr size_t OFF_G1_0 = 0;                       // 256 x  64
constexpr size_t OFF_G1_1 = OFF_G1_0 + 256 * 64;     // 256 x 256
constexpr size_t OFF_G1_2 = OFF_G1_1 + 256 * 256;
constexpr size_t OFF_G1_3 = OFF_G1_2 + 256 * 256;
constexpr size_t OFF_G1_4 = OFF_G1_3 + 256 * 256;
constexpr size_t OFF_G2_0 = OFF_G1_4 + 256 * 256;    // 256 x 320
constexpr size_t OFF_G2_1 = OFF_G2_0 + 256 * 320;
constexpr size_t OFF_G2_2 = OFF_G2_1 + 256 * 256;
constexpr size_t OFF_C_0  = OFF_G2_2 + 256 * 256;    // 256 x 320
constexpr size_t OFF_C_1  = OFF_C_0  + 256 * 320;    //  16 x 256
constexpr size_t OFF_SIG  = OFF_C_1  + 16 * 256;     //  16 x 256
constexpr size_t WS_HALVES = OFF_SIG + 16 * 256;     // ~1.11 MB of f16

// ---------------- prep: fp32 W[din][dout] -> f16 WT[npad][Kpad] ----------------
__global__ void prep_weights(const float* __restrict__ W, _Float16* __restrict__ dst,
                             int din, int dout, int part1, int pad1, int Kpad, int total) {
  int idx = blockIdx.x * blockDim.x + threadIdx.x;
  if (idx >= total) return;
  int n = idx / Kpad;
  int k = idx - n * Kpad;
  float v = 0.f;
  if (n < dout) {
    int r = -1;
    if (k < part1) r = k;
    else if (k >= pad1) {
      int r2 = part1 + (k - pad1);
      if (r2 < din) r = r2;
    }
    if (r >= 0) v = W[(size_t)r * dout + n];
  }
  dst[idx] = (_Float16)v;
}

// ---------------- WMMA fragment helpers ----------------
// A-operand (weights), 16x32 f16 ISA layout: lane m = lane&15, half h = lane>>4;
// elems 0..7 = K(kt*32 + 8h + 0..7), elems 8..15 = K(kt*32 + 16 + 8h + 0..7).
// WT row-major => two global_load_b128 at +0 and +16 halves (base includes +8h).
__device__ __forceinline__ v16h loadAfrag(const _Float16* p) {
  v8h lo = *(const v8h*)p;
  v8h hi = *(const v8h*)(p + 16);
  v16h r;
#pragma unroll
  for (int i = 0; i < 8; ++i) { r[i] = lo[i]; r[i + 8] = hi[i]; }
  return r;
}
// B-operand (activations), 32x16 f16 ISA layout: lane holds col n = lane&15,
// elems 0..7 = K(kt*32 + 8h + 0..7), elems 8..15 = K(kt*32 + 16 + 8h + 0..7) => two ds_load_b128.
__device__ __forceinline__ v16h loadBfrag(const _Float16* p) {
  v8h lo = *(const v8h*)p;
  v8h hi = *(const v8h*)(p + 16);
  v16h r;
#pragma unroll
  for (int i = 0; i < 8; ++i) { r[i] = lo[i]; r[i + 8] = hi[i]; }
  return r;
}

// Full 256-wide layer for this wave's 16 points, transposed orientation:
//   D[M=outfeat][N=point] = WT(A) x H(B).
// Activation B-fragments are loaded once and reused for all 16 output tiles.
// Epilogue: bias (vector f32x4 loads) + optional ReLU + f16 pack + ONE
// ds_store_b128 per tile (8 consecutive halves of the point-major buffer).
template <int KT1, int KT2, bool RELU>
__device__ __forceinline__ void layerT(const _Float16* a1, int s1,
                                       const _Float16* a2, int s2,
                                       const _Float16* __restrict__ WT, int Kpad,
                                       const float* __restrict__ bias,
                                       _Float16* dst, int lo, int h) {
  constexpr int KT = KT1 + KT2;
  v16h bfrag[KT];
  {
    const _Float16* p1 = a1 + lo * s1 + h * 8;
#pragma unroll
    for (int k = 0; k < KT1; ++k) bfrag[k] = loadBfrag(p1 + k * 32);
    if (KT2 > 0) {
      const _Float16* p2 = a2 + lo * s2 + h * 8;
#pragma unroll
      for (int k = 0; k < KT2; ++k) bfrag[KT1 + k] = loadBfrag(p2 + k * 32);
    }
  }
  for (int mT = 0; mT < 16; ++mT) {
    const _Float16* wrow = WT + (size_t)(mT * 16 + lo) * Kpad + h * 8;
    v4f b0 = *(const v4f*)(bias + mT * 16 + h * 8);
    v4f b1 = *(const v4f*)(bias + mT * 16 + h * 8 + 4);
    v8f acc = {};
#pragma unroll
    for (int k = 0; k < KT; ++k) {
      v16h a = loadAfrag(wrow + k * 32);
      acc = __builtin_amdgcn_wmma_f32_16x16x32_f16(false, a, false, bfrag[k],
                                                   (short)0, acc, false, false);
    }
    v8h pk;
#pragma unroll
    for (int v = 0; v < 8; ++v) {
      float val = acc[v] + (v < 4 ? b0[v] : b1[v - 4]);
      if (RELU) val = fmaxf(val, 0.f);
      pk[v] = (_Float16)val;
    }
    // lane (lo,h): point lo, out-features mT*16 + 8h + 0..7 -> contiguous
    *(v8h*)(dst + lo * 264 + mT * 16 + h * 8) = pk;
  }
}

// Single 16-wide head tile (sigma / rgb): same orientation, raw accumulators.
template <int KT>
__device__ __forceinline__ v8f outtileT(const _Float16* src, int s,
                                        const _Float16* __restrict__ WT, int Kpad,
                                        int lo, int h) {
  const _Float16* p = src + lo * s + h * 8;
  const _Float16* wrow = WT + (size_t)lo * Kpad + h * 8;
  v8f acc = {};
#pragma unroll
  for (int k = 0; k < KT; ++k) {
    v16h a = loadAfrag(wrow + k * 32);
    v16h b = loadBfrag(p + k * 32);
    acc = __builtin_amdgcn_wmma_f32_16x16x32_f16(false, a, false, b,
                                                 (short)0, acc, false, false);
  }
  return acc;
}

// ---------------- main fused NeRF kernel: 1 wave = 16 points ----------------
__global__ __launch_bounds__(32) void nerf_forward(
    const float* __restrict__ x, const float* __restrict__ dir,
    const _Float16* __restrict__ wts,
    const float* __restrict__ b_g10, const float* __restrict__ b_g11,
    const float* __restrict__ b_g12, const float* __restrict__ b_g13,
    const float* __restrict__ b_g14,
    const float* __restrict__ b_g20, const float* __restrict__ b_g21,
    const float* __restrict__ b_g22,
    const float* __restrict__ b_c0, const float* __restrict__ b_c1,
    const float* __restrict__ b_sig,
    float* __restrict__ out) {
  __shared__ __align__(16) _Float16 sPE[16 * 72];   // pe(x), 63 -> pad 64
  __shared__ __align__(16) _Float16 sPD[16 * 72];   // pe(dir), 39 -> pad 64
  __shared__ __align__(16) _Float16 sH0[16 * 264];  // ping (point-major)
  __shared__ __align__(16) _Float16 sH1[16 * 264];  // pong

  const int lane = threadIdx.x;
  const int lo = lane & 15;
  const int h = lane >> 4;
  const int rowBase = blockIdx.x * 16;

  // ---- positional encoding: lane (lo,h) fills point lo, cols [h*32, h*32+32) ----
  {
    const int row = rowBase + lo;
    float xv[3], dv[3];
#pragma unroll
    for (int c = 0; c < 3; ++c) {
      xv[c] = x[row * 3 + c];
      dv[c] = dir[row * 3 + c];
    }
    for (int j = h * 32; j < h * 32 + 32; ++j) {
      float v = 0.f, w = 0.f;
      if (j < 3) {
        v = xv[j];
        w = dv[j];
      } else {
        int jj = j - 3;
        int l = jj / 6;
        int st = (jj % 6) / 3;  // 0=sin, 1=cos
        int c = jj % 3;
        float fs = (float)(1 << l);
        if (jj < 60) v = st ? __cosf(xv[c] * fs) : __sinf(xv[c] * fs);  // L_LOC=10
        if (jj < 36) w = st ? __cosf(dv[c] * fs) : __sinf(dv[c] * fs);  // L_DIR=6
      }
      sPE[lo * 72 + j] = (_Float16)v;
      sPD[lo * 72 + j] = (_Float16)w;
    }
  }
  __syncthreads();  // single-wave WG: barrier collapses, keeps DS ordering explicit

  // ---- g1: 63(->64) -> 256 x5 (last no ReLU) ----
  layerT<2, 0, true >(sPE, 72, nullptr, 0, wts + OFF_G1_0, 64,  b_g10, sH0, lo, h);
  layerT<8, 0, true >(sH0, 264, nullptr, 0, wts + OFF_G1_1, 256, b_g11, sH1, lo, h);
  layerT<8, 0, true >(sH1, 264, nullptr, 0, wts + OFF_G1_2, 256, b_g12, sH0, lo, h);
  layerT<8, 0, true >(sH0, 264, nullptr, 0, wts + OFF_G1_3, 256, b_g13, sH1, lo, h);
  layerT<8, 0, false>(sH1, 264, nullptr, 0, wts + OFF_G1_4, 256, b_g14, sH0, lo, h);
  // ---- g2: concat(pe_x, feat1) = 320 -> 256 x3 (last no ReLU) ----
  layerT<2, 8, true >(sPE, 72, sH0, 264, wts + OFF_G2_0, 320, b_g20, sH1, lo, h);
  layerT<8, 0, true >(sH1, 264, nullptr, 0, wts + OFF_G2_1, 256, b_g21, sH0, lo, h);
  layerT<8, 0, false>(sH0, 264, nullptr, 0, wts + OFF_G2_2, 256, b_g22, sH1, lo, h);
  // feat2 in sH1

  // ---- sigma head: feat2 @ sig_W (256 -> 1, padded to 16 out-features) ----
  v8f accS = outtileT<8>(sH1, 264, wts + OFF_SIG, 256, lo, h);
  // ---- color: concat(pe_d, feat2) = 320 -> 256 (ReLU) -> 3 (padded 16) ----
  layerT<2, 8, true>(sPD, 72, sH1, 264, wts + OFF_C_0, 320, b_c0, sH0, lo, h);
  v8f accC = outtileT<8>(sH0, 264, wts + OFF_C_1, 256, lo, h);

  // ---- write [r, g, b, sigma] as one float4 per point (h==0 lanes hold M=0..7) ----
  if (h == 0) {
    v4f o;
    o[0] = accC[0] + b_c1[0];
    o[1] = accC[1] + b_c1[1];
    o[2] = accC[2] + b_c1[2];
    o[3] = accS[0] + b_sig[0];
    *(v4f*)(out + (size_t)(rowBase + lo) * 4) = o;
  }
}

extern "C" void kernel_launch(void* const* d_in, const int* in_sizes, int n_in,
                              void* d_out, int out_size, void* d_ws, size_t ws_size,
                              hipStream_t stream) {
  _Float16* wts = (_Float16*)d_ws;  // needs WS_HALVES*2 ~ 1.11 MB

  struct LD { int wi, din, dout, part1, pad1, Kpad, total; size_t off; };
  const LD layers[11] = {
      { 2,  63, 256,  63,  64,  64,  64 * 256, OFF_G1_0},
      { 4, 256, 256, 256, 256, 256, 256 * 256, OFF_G1_1},
      { 6, 256, 256, 256, 256, 256, 256 * 256, OFF_G1_2},
      { 8, 256, 256, 256, 256, 256, 256 * 256, OFF_G1_3},
      {10, 256, 256, 256, 256, 256, 256 * 256, OFF_G1_4},
      {12, 319, 256,  63,  64, 320, 320 * 256, OFF_G2_0},
      {14, 256, 256, 256, 256, 256, 256 * 256, OFF_G2_1},
      {16, 256, 256, 256, 256, 256, 256 * 256, OFF_G2_2},
      {18, 295, 256,  39,  64, 320, 320 * 256, OFF_C_0},
      {20, 256,   3, 256, 256, 256, 256 * 16,  OFF_C_1},
      {22, 256,   1, 256, 256, 256, 256 * 16,  OFF_SIG},
  };
  for (int i = 0; i < 11; ++i) {
    const LD& L = layers[i];
    int blocks = (L.total + 255) / 256;
    prep_weights<<<blocks, 256, 0, stream>>>((const float*)d_in[L.wi], wts + L.off,
                                             L.din, L.dout, L.part1, L.pad1, L.Kpad,
                                             L.total);
  }

  int n = in_sizes[0] / 3;  // 262144 points
  nerf_forward<<<n / 16, 32, 0, stream>>>(
      (const float*)d_in[0], (const float*)d_in[1], wts,
      (const float*)d_in[3], (const float*)d_in[5], (const float*)d_in[7],
      (const float*)d_in[9], (const float*)d_in[11],
      (const float*)d_in[13], (const float*)d_in[15], (const float*)d_in[17],
      (const float*)d_in[19], (const float*)d_in[21], (const float*)d_in[23],
      (float*)d_out);
}
